// Char_GRU_83519934038526
// MI455X (gfx1250) — compile-verified, matrix operand order checked by reading, not run on hardware
//
#include <hip/hip_runtime.h>
#include <hip/hip_bf16.h>
#include <cstdint>

// ---------------------------------------------------------------------------
// Types / helpers
// ---------------------------------------------------------------------------
typedef __attribute__((ext_vector_type(16))) __bf16 bf16x16;
typedef __attribute__((ext_vector_type(8)))  float  f32x8;
typedef __attribute__((ext_vector_type(4)))  unsigned int u32x4;
typedef __attribute__((ext_vector_type(8)))  int i32x8;
typedef __attribute__((ext_vector_type(4)))  int i32x4;

#if defined(__has_builtin)
#if __has_builtin(__builtin_amdgcn_tensor_load_to_lds)
#define HAVE_TDM 1
#endif
#endif

__device__ inline unsigned short f2bf(float f) {
  union { __bf16 h; unsigned short s; } c; c.h = (__bf16)f; return c.s;
}
__device__ inline float bf2f(unsigned short v) {
  union { __bf16 h; unsigned short s; } c; c.s = v; return (float)c.h;
}
__device__ inline float sigmoid_f(float x) { return 1.0f / (1.0f + expf(-x)); }

union FragU { bf16x16 v; u32x4 q[2]; };

// Async global->LDS 16B copy (CDNA5 GLOBAL_LOAD_ASYNC_TO_LDS_B128, ASYNCcnt).
__device__ inline void async_ld_b128(unsigned lds_addr, const void* g) {
  asm volatile("global_load_async_to_lds_b128 %0, %1, off"
               :: "v"(lds_addr), "v"((unsigned long long)(uintptr_t)g)
               : "memory");
}
__device__ inline void wait_async0() {
  asm volatile("s_wait_asynccnt 0x0" ::: "memory");
}
__device__ inline unsigned lds_addr_of(const void* p) {
  return (unsigned)(uintptr_t)p;  // LDS aperture: offset lives in addr[31:0]
}

// Load a 16x32 bf16 fragment (A layout) or 32x16 (B layout, mirrored) from a
// row-major [rows][stride] bf16 buffer. Per CDNA5 ISA 7.12.2: lanes 0-15 hold
// row (lane), K runs {0..7,16..23}; lanes 16-31 hold K runs {8..15,24..31}.
__device__ inline bf16x16 load_frag(const unsigned short* p, int stride,
                                    int row_base, int k_base) {
  int l = threadIdx.x & 31;
  int r  = row_base + (l & 15);
  int ko = k_base + ((l >> 4) << 3);
  FragU f;
  f.q[0] = *(const u32x4*)(p + (size_t)r * stride + ko);
  f.q[1] = *(const u32x4*)(p + (size_t)r * stride + ko + 16);
  return f.v;
}

__device__ inline f32x8 wmma_bf(bf16x16 a, bf16x16 b, f32x8 c) {
  return __builtin_amdgcn_wmma_f32_16x16x32_bf16(false, a, false, b,
                                                 (short)0, c, false, false);
}

// Device-scope barrier for a small cooperative grid (monotonic counter).
__device__ inline void grid_barrier(unsigned int* cnt, unsigned int gen, int nblocks) {
  __threadfence();
  __syncthreads();
  if (threadIdx.x == 0) {
    __hip_atomic_fetch_add(cnt, 1u, __ATOMIC_ACQ_REL, __HIP_MEMORY_SCOPE_AGENT);
    unsigned int target = gen * (unsigned int)nblocks;
    while (__hip_atomic_load(cnt, __ATOMIC_ACQUIRE, __HIP_MEMORY_SCOPE_AGENT) < target) {
      __builtin_amdgcn_s_sleep(2);
    }
  }
  __syncthreads();
}

// ---------------------------------------------------------------------------
// f32 -> bf16 conversion
// ---------------------------------------------------------------------------
__global__ void __launch_bounds__(256) f2bf_kernel(const float* __restrict__ s,
                                                   unsigned short* __restrict__ d, int n) {
  int i = blockIdx.x * 256 + threadIdx.x;
  if (i < n) d[i] = f2bf(s[i]);
}

// ---------------------------------------------------------------------------
// Build x_input = [embedding[x] , ctx @ W_ctx^T + b_ctx]  (bf16, M x 256)
// ---------------------------------------------------------------------------
__global__ void __launch_bounds__(128) build_input_kernel(
    const int* __restrict__ x, const float* __restrict__ ctx,
    const float* __restrict__ emb, const float* __restrict__ Wctx,
    const float* __restrict__ bctx, unsigned short* __restrict__ xin) {
  __shared__ float cs[300];
  int tok = blockIdx.x, tid = threadIdx.x;
  for (int j = tid; j < 300; j += 128) cs[j] = ctx[(size_t)tok * 300 + j];
  __syncthreads();
  float a = bctx[tid];
  for (int j = 0; j < 300; ++j) a += Wctx[tid * 300 + j] * cs[j];
  xin[(size_t)tok * 256 + 128 + tid] = f2bf(a);
  int id = x[tok];
  xin[(size_t)tok * 256 + tid] = f2bf(emb[id * 128 + tid]);
}

// ---------------------------------------------------------------------------
// Generic bf16 GEMM:  C[M,N] = act(A[M,K] @ B[N,K]^T + bias)
// Block tile 128x64, 8 waves (4x2), each wave 32x32 via 2x2 WMMA frags.
// Tiles staged with async global->LDS; next-K tiles prefetched.
// ---------------------------------------------------------------------------
template<bool OUT_F32, bool TANH_ACT>
__global__ void __launch_bounds__(256) gemm_kernel(
    const unsigned short* __restrict__ A, int lda,
    const unsigned short* __restrict__ Bw, int ldb,
    const float* __restrict__ bias,
    void* __restrict__ C, int ldc, int N, int K) {
  __shared__ unsigned short As[128 * 32];
  __shared__ unsigned short Bs[64 * 32];
  int tid = threadIdx.x, lane = tid & 31, wv = tid >> 5;
  int wm = wv & 3, wn = wv >> 2;
  int mBlk = blockIdx.x * 128, nBlk = blockIdx.y * 64;

  f32x8 acc[2][2];
  for (int i = 0; i < 2; ++i)
    for (int j = 0; j < 2; ++j)
      for (int v = 0; v < 8; ++v) acc[i][j][v] = 0.0f;

  int kt_end = K >> 5;
  for (int kt = 0; kt < kt_end; ++kt) {
    for (int i = 0; i < 2; ++i) {
      int cid = tid + i * 256;
      int r = cid >> 2, k8 = (cid & 3) << 3;
      async_ld_b128(lds_addr_of(As + r * 32 + k8),
                    A + (size_t)(mBlk + r) * lda + kt * 32 + k8);
    }
    {
      int r = tid >> 2, k8 = (tid & 3) << 3;
      int n = nBlk + r;
      if (n < N) {
        async_ld_b128(lds_addr_of(Bs + r * 32 + k8),
                      Bw + (size_t)n * ldb + kt * 32 + k8);
      } else {
        u32x4 z = {0u, 0u, 0u, 0u};
        *(u32x4*)(Bs + r * 32 + k8) = z;
      }
    }
    if (kt + 1 < kt_end) {  // prefetch next K tiles (global_prefetch_b8)
      __builtin_prefetch(A + (size_t)(mBlk + (tid & 127)) * lda + (kt + 1) * 32, 0, 1);
      if (tid < 64 && nBlk + tid < N)
        __builtin_prefetch(Bw + (size_t)(nBlk + tid) * ldb + (kt + 1) * 32, 0, 1);
    }
    wait_async0();
    __syncthreads();
    bf16x16 af[2], bfv[2];
    af[0]  = load_frag(As, 32, wm * 32, 0);
    af[1]  = load_frag(As, 32, wm * 32 + 16, 0);
    bfv[0] = load_frag(Bs, 32, wn * 32, 0);
    bfv[1] = load_frag(Bs, 32, wn * 32 + 16, 0);
    for (int i = 0; i < 2; ++i)
      for (int j = 0; j < 2; ++j)
        acc[i][j] = wmma_bf(af[i], bfv[j], acc[i][j]);
    __syncthreads();
  }

  for (int i = 0; i < 2; ++i)
    for (int j = 0; j < 2; ++j)
      for (int v = 0; v < 8; ++v) {
        int row = mBlk + wm * 32 + i * 16 + v + 8 * (lane >> 4);
        int col = nBlk + wn * 32 + j * 16 + (lane & 15);
        if (col < N) {
          float val = acc[i][j][v];
          if (bias) val += bias[col];
          if (TANH_ACT) val = tanhf(val);
          if (OUT_F32) ((float*)C)[(size_t)row * ldc + col] = val;
          else ((unsigned short*)C)[(size_t)row * ldc + col] = f2bf(val);
        }
      }
}

// ---------------------------------------------------------------------------
// Persistent cooperative GRU recurrence. 16 WGs x 256 threads; WG `wg` owns
// H-columns [wg*32, wg*32+32). Wh slice (96x512 bf16) pinned in LDS via the
// Tensor Data Mover (3 x 2D tiles of 32x512). Per-step h staged with async
// global->LDS loads. Waves 0-1: r, 2-3: z, 4-5: n + state update.
// ---------------------------------------------------------------------------
__global__ void __launch_bounds__(256) gru_seq_kernel(
    const unsigned short* __restrict__ Wh_bf,   // 1536 x 512 bf16
    const float* __restrict__ bh,               // 1536
    const unsigned short* __restrict__ xi,      // (B*L) x 1536 bf16 (= x@Wi^T+bi)
    unsigned short* __restrict__ ybuf,          // (B*L) x 512 bf16 out
    const float* __restrict__ h0,               // 16 x 512
    float* __restrict__ hbuf,                   // 2 x 16 x 512 f32
    unsigned short* __restrict__ hbf,           // 2 x 16 x 512 bf16
    float* __restrict__ hout,                   // 16 x 512 (final h)
    unsigned int* __restrict__ bar,
    int L, int nblocks) {
  extern __shared__ char smem[];
  unsigned short* WhS = (unsigned short*)smem;      // 96*512
  unsigned short* hS  = WhS + 96 * 512;             // 16*512
  float* rS = (float*)(hS + 16 * 512);              // [2][16][16]
  float* zS = rS + 512;                             // [2][16][16]

  const int tid = threadIdx.x, lane = tid & 31, wv = tid >> 5;
  const int wg = blockIdx.x;
  const int g = wv >> 1, nt = wv & 1;
  const int col16 = lane & 15, hi = lane >> 4;

#ifdef HAVE_TDM
  // Pin Wh rows {gg*512 + wg*32 .. +32} via TDM: tensor (1536x512, 2B elems),
  // tile 32x512 at the tile-start address, contiguous into LDS.
  if (wv == 0) {
    for (int gg = 0; gg < 3; ++gg) {
      unsigned long long ga =
          (unsigned long long)(uintptr_t)(Wh_bf + ((size_t)gg * 512 + wg * 32) * 512);
      u32x4 g0;
      g0[0] = 1u;                                   // count=1, user descriptor
      g0[1] = lds_addr_of(WhS + gg * 32 * 512);     // lds_addr
      g0[2] = (unsigned)(ga & 0xffffffffu);         // global_addr[31:0]
      g0[3] = (unsigned)((ga >> 32) & 0x1ffffffu) | (2u << 30);  // [56:32] | type=2
      i32x8 g1;
      g1[0] = 0x10000;                  // data_size=1 (2 bytes), no multicast
      g1[1] = (int)(512u << 16);        // tensor_dim0[15:0]=512
      g1[2] = (int)(1536u << 16);       // tensor_dim1[15:0]=1536
      g1[3] = (int)(512u << 16);        // tile_dim0=512
      g1[4] = 32;                       // tile_dim1=32
      g1[5] = 512;                      // tensor_dim0_stride=512
      g1[6] = 0; g1[7] = 0;
      i32x4 gz; gz[0] = 0; gz[1] = 0; gz[2] = 0; gz[3] = 0;
      i32x8 gz8;
      for (int q = 0; q < 8; ++q) gz8[q] = 0;
      __builtin_amdgcn_tensor_load_to_lds(g0, g1, gz, gz, gz8, 0);
    }
  }
  __builtin_amdgcn_s_wait_tensorcnt(0);
#else
  for (int cid = tid; cid < 96 * 64; cid += 256) {
    int lr = cid >> 6, k8 = (cid & 63) << 3;
    int gg = lr >> 5, c = lr & 31;
    int grow = gg * 512 + wg * 32 + c;
    *(u32x4*)(WhS + lr * 512 + k8) = *(const u32x4*)(Wh_bf + (size_t)grow * 512 + k8);
  }
#endif
  // Seed h state (this WG's column slice).
  for (int i = tid; i < 512; i += 256) {
    int b = i >> 5, c = i & 31;
    int col = wg * 32 + c;
    float v = h0[b * 512 + col];
    hbuf[b * 512 + col] = v;
    hbf[b * 512 + col]  = f2bf(v);
  }
  grid_barrier(bar, 1u, nblocks);

  for (int t = 0; t < L; ++t) {
    int cur = t & 1, nxt = cur ^ 1;
    // Stage full h_t (16x512 bf16) to LDS asynchronously.
    for (int cid = tid; cid < 1024; cid += 256) {
      async_ld_b128(lds_addr_of(hS + cid * 8), hbf + cur * 8192 + cid * 8);
    }
    wait_async0();
    __syncthreads();

    f32x8 acc; for (int v = 0; v < 8; ++v) acc[v] = 0.0f;
    float xv[8], hg[8];
    int colg = wg * 32 + nt * 16 + col16;

    if (wv < 6) {
      // Software-pipelined K loop: fragments for ks+1 issued before WMMA(ks)
      // so LDS reads overlap matrix issue (partial s_wait_dscnt).
      const int rowB = g * 32 + nt * 16;
      bf16x16 a0 = load_frag(hS, 512, 0, 0);
      bf16x16 b0 = load_frag(WhS, 512, rowB, 0);
#pragma unroll
      for (int ks = 0; ks < 15; ++ks) {
        bf16x16 a1 = load_frag(hS, 512, 0, (ks + 1) * 32);
        bf16x16 b1 = load_frag(WhS, 512, rowB, (ks + 1) * 32);
        acc = wmma_bf(a0, b0, acc);
        a0 = a1; b0 = b1;
      }
      acc = wmma_bf(a0, b0, acc);

      float bhv = bh[g * 512 + colg];
      for (int v = 0; v < 8; ++v) {
        int b = v + 8 * hi;
        xv[v] = bf2f(xi[((size_t)(b * L + t)) * 1536 + g * 512 + colg]);
        hg[v] = acc[v] + bhv;   // (h @ Wh^T + bh) element
      }
      if (t + 1 < L) {  // prefetch next step's xi gather lines
        __builtin_prefetch(xi + ((size_t)((hi * 8) * L + t + 1)) * 1536 + g * 512 + colg, 0, 1);
      }
      if (g == 0) {
        for (int v = 0; v < 8; ++v)
          rS[nt * 256 + (v + 8 * hi) * 16 + col16] = sigmoid_f(xv[v] + hg[v]);
      } else if (g == 1) {
        for (int v = 0; v < 8; ++v)
          zS[nt * 256 + (v + 8 * hi) * 16 + col16] = sigmoid_f(xv[v] + hg[v]);
      }
    }
    __syncthreads();

    if (wv >= 4 && wv < 6) {
      for (int v = 0; v < 8; ++v) {
        int b = v + 8 * hi;
        float r = rS[nt * 256 + b * 16 + col16];
        float z = zS[nt * 256 + b * 16 + col16];
        float n = tanhf(xv[v] + r * hg[v]);
        float ho = hbuf[cur * 8192 + b * 512 + colg];
        float hn = (1.0f - z) * n + z * ho;
        hbuf[nxt * 8192 + b * 512 + colg] = hn;
        hbf[nxt * 8192 + b * 512 + colg]  = f2bf(hn);
        ybuf[((size_t)(b * L + t)) * 512 + colg] = f2bf(hn);
        if (t == L - 1) hout[b * 512 + colg] = hn;
      }
    }
    grid_barrier(bar, (unsigned int)(t + 2), nblocks);
  }
}

// ---------------------------------------------------------------------------
// LayerNorm over H=512; writes bf16 into cat[:, 0:512] (ld = 1024).
// ---------------------------------------------------------------------------
__global__ void __launch_bounds__(256) ln_kernel(
    const unsigned short* __restrict__ y, const float* __restrict__ g,
    const float* __restrict__ b, unsigned short* __restrict__ cat) {
  __shared__ float red[256];
  int tok = blockIdx.x, tid = threadIdx.x;
  float v0 = bf2f(y[(size_t)tok * 512 + tid]);
  float v1 = bf2f(y[(size_t)tok * 512 + tid + 256]);
  red[tid] = v0 + v1;
  __syncthreads();
  for (int st = 128; st > 0; st >>= 1) {
    if (tid < st) red[tid] += red[tid + st];
    __syncthreads();
  }
  float mu = red[0] * (1.0f / 512.0f);
  __syncthreads();
  float d0 = v0 - mu, d1 = v1 - mu;
  red[tid] = d0 * d0 + d1 * d1;
  __syncthreads();
  for (int st = 128; st > 0; st >>= 1) {
    if (tid < st) red[tid] += red[tid + st];
    __syncthreads();
  }
  float rstd = rsqrtf(red[0] * (1.0f / 512.0f) + 1e-5f);
  cat[(size_t)tok * 1024 + tid]       = f2bf(d0 * rstd * g[tid] + b[tid]);
  cat[(size_t)tok * 1024 + tid + 256] = f2bf(d1 * rstd * g[tid + 256] + b[tid + 256]);
}

// ---------------------------------------------------------------------------
// Causal flash attention. One WG per (q-tile of 16, batch). 8 waves split H
// (64 cols each). S computed split-K with ds_add_f32 reduction in LDS.
// ---------------------------------------------------------------------------
__global__ void __launch_bounds__(256) attn_kernel(
    const unsigned short* __restrict__ cat,   // (B*L) x 1024; [0:512] = LN out
    const unsigned short* __restrict__ keys,  // (B*L) x 512
    unsigned short* __restrict__ catw,        // ctx written to [512:1024]
    int L) {
  __shared__ unsigned short Qs[16 * 512];
  __shared__ unsigned short Vt[512 * 32];     // transposed V chunk [hcol][key]
  __shared__ float S[16 * 32];
  __shared__ unsigned short Ps[16 * 32];
  __shared__ float mS[16], lS[16], aS[16];

  int qt = blockIdx.x, b = blockIdx.y;
  int tid = threadIdx.x, lane = tid & 31, wv = tid >> 5;
  size_t base = (size_t)b * L;

  for (int cid = tid; cid < 1024; cid += 256) {
    int q = cid >> 6, h8 = (cid & 63) << 3;
    async_ld_b128(lds_addr_of(Qs + q * 512 + h8),
                  cat + (base + qt * 16 + q) * 1024 + h8);
  }
  wait_async0();
  if (tid < 16) { mS[tid] = -1e30f; lS[tid] = 0.0f; }

  f32x8 o[4];
  for (int i = 0; i < 4; ++i)
    for (int v = 0; v < 8; ++v) o[i][v] = 0.0f;

  int nchunks = (qt * 16 + 47) >> 5;
  const float scale = 0.04419417382415922f;  // 1/sqrt(512)

  for (int kc = 0; kc < nchunks; ++kc) {
    int kbase = kc * 32;
    __syncthreads();  // previous iteration's Vt/Ps readers done
    // Stage V chunk transposed, zero S.
    for (int cid = tid; cid < 2048; cid += 256) {
      int k = cid >> 6, h8 = (cid & 63) << 3;
      u32x4 d = *(const u32x4*)(cat + (base + kbase + k) * 1024 + h8);
      unsigned short tmp[8];
      *(u32x4*)tmp = d;
      for (int e = 0; e < 8; ++e) Vt[(h8 + e) * 32 + k] = tmp[e];
    }
    if (kc + 1 < nchunks && tid < 32) {  // prefetch next V chunk rows
      __builtin_prefetch(cat + (base + kbase + 32 + tid) * 1024, 0, 1);
      __builtin_prefetch(keys + (base + kbase + 32 + tid) * 512, 0, 1);
    }
    for (int i = tid; i < 512; i += 256) S[i] = 0.0f;
    __syncthreads();
    // Partial S over this wave's 64-col H slice; reduce with LDS float atomics.
    for (int nt = 0; nt < 2; ++nt) {
      f32x8 acc; for (int v = 0; v < 8; ++v) acc[v] = 0.0f;
      for (int ks = 0; ks < 2; ++ks) {
        bf16x16 a  = load_frag(Qs, 512, 0, wv * 64 + ks * 32);
        bf16x16 bk = load_frag(keys + base * 512, 512, kbase + nt * 16, wv * 64 + ks * 32);
        acc = wmma_bf(a, bk, acc);
      }
      for (int v = 0; v < 8; ++v) {
        int r = v + 8 * (lane >> 4);
        atomicAdd(&S[r * 32 + nt * 16 + (lane & 15)], acc[v]);
      }
    }
    __syncthreads();
    // Online softmax update (one thread per query row).
    if (tid < 16) {
      int r = tid, qg = qt * 16 + r;
      float mo = mS[r], mx = mo;
      float sv[32];
      for (int c = 0; c < 32; ++c) {
        float s = S[r * 32 + c] * scale;
        if (kbase + c > qg) s = -1e30f;
        sv[c] = s;
        if (s > mx) mx = s;
      }
      float al = expf(mo - mx);
      float ssum = 0.0f;
      for (int c = 0; c < 32; ++c) {
        float p = expf(sv[c] - mx);
        Ps[r * 32 + c] = f2bf(p);
        ssum += p;
      }
      lS[r] = lS[r] * al + ssum;
      mS[r] = mx;
      aS[r] = al;
    }
    __syncthreads();
    // Rescale O and accumulate P @ V for this wave's H slice.
    {
      bf16x16 a = load_frag(Ps, 32, 0, 0);
      for (int nt2 = 0; nt2 < 4; ++nt2) {
        for (int v = 0; v < 8; ++v) o[nt2][v] *= aS[v + 8 * (lane >> 4)];
        bf16x16 bv = load_frag(Vt, 32, wv * 64 + nt2 * 16, 0);
        o[nt2] = wmma_bf(a, bv, o[nt2]);
      }
    }
  }
  __syncthreads();
  for (int nt2 = 0; nt2 < 4; ++nt2)
    for (int v = 0; v < 8; ++v) {
      int r = v + 8 * (lane >> 4);
      int col = wv * 64 + nt2 * 16 + (lane & 15);
      float val = o[nt2][v] / lS[r];
      catw[(base + qt * 16 + r) * 1024 + 512 + col] = f2bf(val);
    }
}

// ---------------------------------------------------------------------------
// Host launcher
// ---------------------------------------------------------------------------
extern "C" void kernel_launch(void* const* d_in, const int* in_sizes, int n_in,
                              void* d_out, int out_size, void* d_ws, size_t ws_size,
                              hipStream_t stream) {
  const int B = 16, L = 2048, E = 128, H = 512, V = 72;
  const int M = B * L;  // 32768

  const int*   x     = (const int*)d_in[0];
  const float* ctx   = (const float*)d_in[1];
  const float* hin   = (const float*)d_in[2];
  const float* emb   = (const float*)d_in[3];
  const float* Wctx  = (const float*)d_in[4];
  const float* bctx  = (const float*)d_in[5];
  const float* Wi[3] = {(const float*)d_in[6],  (const float*)d_in[10], (const float*)d_in[14]};
  const float* Wh[3] = {(const float*)d_in[7],  (const float*)d_in[11], (const float*)d_in[15]};
  const float* bi[3] = {(const float*)d_in[8],  (const float*)d_in[12], (const float*)d_in[16]};
  const float* bh[3] = {(const float*)d_in[9],  (const float*)d_in[13], (const float*)d_in[17]};
  const float* lng   = (const float*)d_in[18];
  const float* lnb   = (const float*)d_in[19];
  const float* Wa    = (const float*)d_in[20];
  const float* Wcomb = (const float*)d_in[21];
  const float* bcomb = (const float*)d_in[22];
  const float* Wh2e  = (const float*)d_in[23];
  const float* bh2e  = (const float*)d_in[24];

  float* out  = (float*)d_out;
  float* hout = out + (size_t)M * V;  // (3, B, H)

  char* ws = (char*)d_ws;
  size_t off = 0;
  auto alloc = [&](size_t bytes) -> void* {
    off = (off + 255) & ~(size_t)255;
    void* p = ws + off;
    off += bytes;
    return p;
  };
  typedef unsigned short u16;
  u16* emb_w   = (u16*)alloc((size_t)V * E * 2);
  u16* Wi_w0   = (u16*)alloc((size_t)3 * H * 256 * 2);
  u16* Wi_w1   = (u16*)alloc((size_t)3 * H * H * 2);
  u16* Wi_w2   = (u16*)alloc((size_t)3 * H * H * 2);
  u16* Wh_w0   = (u16*)alloc((size_t)3 * H * H * 2);
  u16* Wh_w1   = (u16*)alloc((size_t)3 * H * H * 2);
  u16* Wh_w2   = (u16*)alloc((size_t)3 * H * H * 2);
  u16* Wa_w    = (u16*)alloc((size_t)H * H * 2);
  u16* Wcomb_w = (u16*)alloc((size_t)H * 2 * H * 2);
  u16* Wh2e_w  = (u16*)alloc((size_t)E * H * 2);
  u16* xin     = (u16*)alloc((size_t)M * 256 * 2);
  u16* gbuf    = (u16*)alloc((size_t)M * 1536 * 2);
  u16* ybuf    = (u16*)alloc((size_t)M * 512 * 2);
  u16* cat     = (u16*)alloc((size_t)M * 1024 * 2);
  u16* keysb   = (u16*)alloc((size_t)M * 512 * 2);
  u16* comb    = (u16*)alloc((size_t)M * 512 * 2);
  u16* embs    = (u16*)alloc((size_t)M * 128 * 2);
  float* hbuf  = (float*)alloc((size_t)2 * 16 * 512 * 4);
  u16*   hbf   = (u16*)alloc((size_t)2 * 16 * 512 * 2);
  unsigned int* bars = (unsigned int*)alloc(3 * 256);
  (void)ws_size; (void)in_sizes; (void)n_in; (void)out_size;

  (void)hipMemsetAsync(bars, 0, 3 * 256, stream);

  auto conv = [&](const float* s, u16* d, int n) {
    f2bf_kernel<<<(n + 255) / 256, 256, 0, stream>>>(s, d, n);
  };
  conv(emb, emb_w, V * E);
  conv(Wi[0], Wi_w0, 3 * H * 256);
  conv(Wi[1], Wi_w1, 3 * H * H);
  conv(Wi[2], Wi_w2, 3 * H * H);
  conv(Wh[0], Wh_w0, 3 * H * H);
  conv(Wh[1], Wh_w1, 3 * H * H);
  conv(Wh[2], Wh_w2, 3 * H * H);
  conv(Wa, Wa_w, H * H);
  conv(Wcomb, Wcomb_w, H * 2 * H);
  conv(Wh2e, Wh2e_w, E * H);

  build_input_kernel<<<M, 128, 0, stream>>>(x, ctx, emb, Wctx, bctx, xin);

  const u16* Wi_w[3] = {Wi_w0, Wi_w1, Wi_w2};
  const u16* Wh_w[3] = {Wh_w0, Wh_w1, Wh_w2};

  constexpr int GRU_LDS = 96 * 512 * 2 + 16 * 512 * 2 + 2 * 512 * 4;  // 118784 B
  (void)hipFuncSetAttribute(reinterpret_cast<const void*>(gru_seq_kernel),
                            hipFuncAttributeMaxDynamicSharedMemorySize, GRU_LDS);

  for (int layer = 0; layer < 3; ++layer) {
    const u16* Ain = (layer == 0) ? xin : ybuf;
    int lda = (layer == 0) ? 256 : 512;
    int K   = (layer == 0) ? 256 : 512;
    gemm_kernel<false, false><<<dim3(M / 128, (3 * H) / 64), 256, 0, stream>>>(
        Ain, lda, Wi_w[layer], K, bi[layer], gbuf, 3 * H, 3 * H, K);
    unsigned int* bar = (unsigned int*)((char*)bars + layer * 256);
    gru_seq_kernel<<<16, 256, GRU_LDS, stream>>>(
        Wh_w[layer], bh[layer], gbuf, ybuf, hin + (size_t)layer * B * H,
        hbuf, hbf, hout + (size_t)layer * B * H, bar, L, 16);
  }

  ln_kernel<<<M, 256, 0, stream>>>(ybuf, lng, lnb, cat);

  gemm_kernel<false, false><<<dim3(M / 128, H / 64), 256, 0, stream>>>(
      cat, 1024, Wa_w, 512, nullptr, keysb, 512, H, 512);

  attn_kernel<<<dim3(L / 16, B), 256, 0, stream>>>(cat, keysb, cat, L);

  gemm_kernel<false, true><<<dim3(M / 128, H / 64), 256, 0, stream>>>(
      cat, 1024, Wcomb_w, 1024, bcomb, comb, 512, H, 1024);

  gemm_kernel<false, false><<<dim3(M / 128, E / 64), 256, 0, stream>>>(
      comb, 512, Wh2e_w, 512, bh2e, embs, 128, E, 512);

  gemm_kernel<true, false><<<dim3(M / 128, 2), 256, 0, stream>>>(
      embs, 128, emb_w, 128, nullptr, out, V, V, 128);
}
